// RicciCurvaturePooling1_71622874628247
// MI455X (gfx1250) — matrix operand order repaired and checked by reading, not compiled, not run on hardware
//
#include <hip/hip_runtime.h>
#include <math.h>

// CDNA5 wave32 WMMA types
typedef __attribute__((ext_vector_type(16))) __bf16 v16bf;
typedef __attribute__((ext_vector_type(8)))  float  v8f;
// TDM descriptor groups (6-arg builtin form: g0, g1, g2, g3, g4, cpol)
typedef __attribute__((ext_vector_type(4))) unsigned int u32x4;
typedef __attribute__((ext_vector_type(8))) int i32x8;
typedef __attribute__((ext_vector_type(4))) int i32x4;

static __device__ inline void atomicMaxF32(float* addr, float val) {
  unsigned int* ua = (unsigned int*)addr;
  unsigned int old = *ua;
  for (;;) {
    if (__uint_as_float(old) >= val) break;
    unsigned int assumed = old;
    old = atomicCAS(ua, assumed, __float_as_uint(val));
    if (old == assumed) break;
  }
}

// ---- Tensor Data Mover staging ---------------------------------------------
// Stage a rows x cols fp32 row-major tensor (stride == cols) into LDS with one
// 2D TDM descriptor (CDNA5 ISA). Thread 0 issues the DMA
// and waits on TENSORcnt; the trailing barrier releases the block.
static __device__ inline void tdm_stage_to_lds(const float* gsrc, float* lds,
                                               int cols, int rows) {
  if (threadIdx.x == 0) {
    unsigned long long ga = (unsigned long long)(uintptr_t)gsrc;
    unsigned ldsOff = (unsigned)(uintptr_t)lds;  // generic LDS addr: [31:0] = LDS offset
    u32x4 g0;
    g0[0] = 1u;                                   // count=1, user descriptor
    g0[1] = ldsOff;                               // lds_addr
    g0[2] = (unsigned)ga;                         // global_addr[31:0]
    g0[3] = (unsigned)((ga >> 32) & 0x01FFFFFFu)  // global_addr[56:32]
            | (2u << 30);                         // type = 2 ("image")
    i32x8 g1;
    g1[0] = 2 << 16;                              // data_size = 4 bytes
    g1[1] = (cols & 0xFFFF) << 16;                // tensor_dim0[15:0]   (bits 79:48)
    g1[2] = ((unsigned)cols >> 16) | ((rows & 0xFFFF) << 16); // dim0 hi | dim1 lo
    g1[3] = ((unsigned)rows >> 16) | ((cols & 0xFFFF) << 16); // dim1 hi | tile_dim0
    g1[4] = rows & 0xFFFF;                        // tile_dim1 (tile_dim2 = 0)
    g1[5] = cols;                                 // tensor_dim0_stride[31:0]
    g1[6] = 0;                                    // stride0 hi | stride1 lo
    g1[7] = 0;
    i32x4 z4 = {0, 0, 0, 0};                      // 2D: groups 2/3 unused
    i32x8 z8 = {0, 0, 0, 0, 0, 0, 0, 0};
    __builtin_amdgcn_tensor_load_to_lds(g0, g1, z4, z4, z8, 0);
    __builtin_amdgcn_s_wait_tensorcnt(0);
  }
  __syncthreads();
}

// ---- WMMA fragments (layouts per ISA 7.12.2, wave32) ------------------------
static __device__ inline v16bf load_a_frag(const float* __restrict__ A, int lda,
                                           int mrow, int k0, int half) {
  v16bf a;
#pragma unroll
  for (int v = 0; v < 8; ++v) {
    const int ka = k0 + (v >> 2) * 16 + half * 8 + (v & 3) * 2;
    a[2 * v]     = (__bf16)A[(size_t)mrow * lda + ka];
    a[2 * v + 1] = (__bf16)A[(size_t)mrow * lda + ka + 1];
  }
  return a;
}
static __device__ inline v16bf load_b_frag(const float* __restrict__ B, int ldb,
                                           int k0, int ncol, int half) {
  v16bf b;
#pragma unroll
  for (int v = 0; v < 8; ++v) {
    const int kb = k0 + (v >> 2) * 16 + half * 8 + (v & 3) * 2;
    b[2 * v]     = (__bf16)B[(size_t)kb * ldb + ncol];
    b[2 * v + 1] = (__bf16)B[(size_t)(kb + 1) * ldb + ncol];
  }
  return b;
}

// ---- GEMM kernels -----------------------------------------------------------

// D = A(Nx64) @ B(64x64) [+ bias].  block=128 (4 waves = 4 col tiles).
// B staged to LDS once via TDM; fragments read from LDS (ds_load).
__global__ void k_gemm64(const float* __restrict__ A, const float* __restrict__ B,
                         const float* __restrict__ bias, float* __restrict__ D,
                         int N, int hasBias) {
  __shared__ float Wlds[64 * 64];                 // 16 KB
  tdm_stage_to_lds(B, Wlds, 64, 64);
  const int lane = threadIdx.x & 31;
  const int wave = threadIdx.x >> 5;              // 0..3
  const int half = lane >> 4;
  const int m0 = blockIdx.x * 16;
  const int mrow = min(m0 + (lane & 15), N - 1);
  const int ncol = wave * 16 + (lane & 15);
  v8f acc = {};
#pragma unroll
  for (int k0 = 0; k0 < 64; k0 += 32) {
    v16bf a = load_a_frag(A, 64, mrow, k0, half);
    v16bf b = load_b_frag(Wlds, 64, k0, ncol, half);
    acc = __builtin_amdgcn_wmma_f32_16x16x32_bf16(false, a, false, b,
                                                  (short)0, acc, false, false);
  }
  const float bv = hasBias ? bias[ncol] : 0.0f;
  const int mbase = m0 + half * 8;
  if (mbase + 8 <= N) {
#pragma unroll
    for (int r = 0; r < 8; ++r) D[(size_t)(mbase + r) * 64 + ncol] = acc[r] + bv;
  } else {
    for (int r = 0; r < 8 && mbase + r < N; ++r)
      D[(size_t)(mbase + r) * 64 + ncol] = acc[r] + bv;
  }
}

// D = A(Nx64) @ B(64x384).  block=256 (8 waves x 3 col tiles).
// 32x384 K-slab of B staged to LDS per K step via TDM.
__global__ void k_gemm384(const float* __restrict__ A, const float* __restrict__ B,
                          float* __restrict__ D, int N) {
  __shared__ float Blds[32 * 384];                // 48 KB
  const int lane = threadIdx.x & 31;
  const int wave = threadIdx.x >> 5;              // 0..7
  const int half = lane >> 4;
  const int m0 = blockIdx.x * 16;
  const int mrow = min(m0 + (lane & 15), N - 1);
  v8f acc[3] = {};
  for (int k0 = 0; k0 < 64; k0 += 32) {
    tdm_stage_to_lds(B + (size_t)k0 * 384, Blds, 384, 32);  // barrier inside
    v16bf a = load_a_frag(A, 64, mrow, k0, half);
#pragma unroll
    for (int j = 0; j < 3; ++j) {
      const int ncol = (wave * 3 + j) * 16 + (lane & 15);
      v16bf b = load_b_frag(Blds, 384, 0, ncol, half);
      acc[j] = __builtin_amdgcn_wmma_f32_16x16x32_bf16(false, a, false, b,
                                                       (short)0, acc[j], false, false);
    }
    __syncthreads();                              // protect Blds for next slab
  }
  const int mbase = m0 + half * 8;
#pragma unroll
  for (int j = 0; j < 3; ++j) {
    const int ncol = (wave * 3 + j) * 16 + (lane & 15);
    if (mbase + 8 <= N) {
#pragma unroll
      for (int r = 0; r < 8; ++r) D[(size_t)(mbase + r) * 384 + ncol] = acc[j][r];
    } else {
      for (int r = 0; r < 8 && mbase + r < N; ++r)
        D[(size_t)(mbase + r) * 384 + ncol] = acc[j][r];
    }
  }
}

// ---- degree / GCN aggregation ----------------------------------------------

__global__ void k_init_deg(float* deg, int N) {
  int i = blockIdx.x * blockDim.x + threadIdx.x;
  if (i < N) deg[i] = 1.0f;                       // self loop
}
__global__ void k_deg_accum(const int* __restrict__ dst, float* deg, int E) {
  int e = blockIdx.x * blockDim.x + threadIdx.x;
  if (e < E) atomicAdd(&deg[dst[e]], 1.0f);
}
__global__ void k_deg_to_dinv(float* deg, int N) {
  int i = blockIdx.x * blockDim.x + threadIdx.x;
  if (i < N) deg[i] = rsqrtf(fmaxf(deg[i], 1.0f));
}
__global__ void k_init_xpool(const float* __restrict__ dinv, const float* __restrict__ xw,
                             float* xp, float* xout, int N) {
  int idx = blockIdx.x * blockDim.x + threadIdx.x;
  if (idx < N * 64) {
    float d = dinv[idx >> 6];
    xp[idx] = d * d * xw[idx];
    xout[idx] = 0.0f;
  }
}
// wave per edge: x_pool[dst] += dinv[src]*dinv[dst] * xw[src]
__global__ void k_scatter_xpool(const int* __restrict__ src, const int* __restrict__ dst,
                                const float* __restrict__ dinv, const float* __restrict__ xw,
                                float* xp, int E) {
  int e = blockIdx.x * (blockDim.x >> 5) + (threadIdx.x >> 5);
  int lane = threadIdx.x & 31;
  if (e >= E) return;
  int s = src[e], t = dst[e];
  float w = dinv[s] * dinv[t];
  atomicAdd(&xp[(size_t)t * 64 + lane],      w * xw[(size_t)s * 64 + lane]);
  atomicAdd(&xp[(size_t)t * 64 + lane + 32], w * xw[(size_t)s * 64 + lane + 32]);
}

// ---- attention --------------------------------------------------------------

__global__ void k_init_xjmax(const float* __restrict__ xh, float* xjm, int N) {
  int idx = blockIdx.x * blockDim.x + threadIdx.x;
  if (idx < N * 384) xjm[idx] = xh[idx];          // self loop
}
__global__ void k_init_seg(float* mseg, float* zseg, int N) {
  int idx = blockIdx.x * blockDim.x + threadIdx.x;
  if (idx < N * 6) { mseg[idx] = -INFINITY; zseg[idx] = 0.0f; }
}
// wave per edge: xjmax[src] = max(xjmax[src], xh[dst])  (384 floats)
__global__ void k_segmax(const int* __restrict__ src, const int* __restrict__ dst,
                         const float* __restrict__ xh, float* xjm, int E) {
  int e = blockIdx.x * (blockDim.x >> 5) + (threadIdx.x >> 5);
  int lane = threadIdx.x & 31;
  if (e >= E) return;
  int t = src[e], s = dst[e];
  const float* xs = xh + (size_t)s * 384;
  float* xm = xjm + (size_t)t * 384;
#pragma unroll
  for (int j = 0; j < 12; ++j) atomicMaxF32(&xm[j * 32 + lane], xs[j * 32 + lane]);
}
// wave per (edge incl self-loops): alpha = leakyrelu(dot(xh[s],att_i)+dot(xjm[t],att_j))
__global__ void k_alpha(const int* __restrict__ src, const int* __restrict__ dst,
                        const float* __restrict__ xh, const float* __restrict__ xjm,
                        const float* __restrict__ att, float* alpha, float* mseg,
                        int E, int N) {
  int e = blockIdx.x * (blockDim.x >> 5) + (threadIdx.x >> 5);
  int lane = threadIdx.x & 31;
  if (e >= E + N) return;
  int t, s;
  if (e < E) { t = src[e]; s = dst[e]; } else { t = s = e - E; }
  const float* xs = xh  + (size_t)s * 384;
  const float* xm = xjm + (size_t)t * 384;
#pragma unroll
  for (int h = 0; h < 6; ++h) {
    float p = xs[h * 64 + lane]      * att[h * 128 + lane]
            + xs[h * 64 + lane + 32] * att[h * 128 + lane + 32]
            + xm[h * 64 + lane]      * att[h * 128 + 64 + lane]
            + xm[h * 64 + lane + 32] * att[h * 128 + 64 + lane + 32];
#pragma unroll
    for (int o = 16; o > 0; o >>= 1) p += __shfl_xor(p, o, 32);
    if (lane == 0) {
      float av = p > 0.0f ? p : 0.2f * p;         // leaky relu
      alpha[(size_t)e * 6 + h] = av;
      atomicMaxF32(&mseg[t * 6 + h], av);
    }
  }
}
// thread per (edge,head): exp(alpha - m), accumulate z
__global__ void k_exp(const int* __restrict__ src, float* alpha,
                      const float* __restrict__ mseg, float* zseg, int E, int N) {
  int idx = blockIdx.x * blockDim.x + threadIdx.x;
  if (idx >= (E + N) * 6) return;
  int e = idx / 6, h = idx % 6;
  int t = (e < E) ? src[e] : (e - E);
  float ev = expf(alpha[idx] - mseg[t * 6 + h]);
  alpha[idx] = ev;
  atomicAdd(&zseg[t * 6 + h], ev);
}
// wave per edge: x_out[t] += (1/H) * sum_h (e/z) * xh[s,h,:]
__global__ void k_xout(const int* __restrict__ src, const int* __restrict__ dst,
                       const float* __restrict__ xh, const float* __restrict__ alpha,
                       const float* __restrict__ zseg, float* xout, int E, int N) {
  int e = blockIdx.x * (blockDim.x >> 5) + (threadIdx.x >> 5);
  int lane = threadIdx.x & 31;
  if (e >= E + N) return;
  int t, s;
  if (e < E) { t = src[e]; s = dst[e]; } else { t = s = e - E; }
  const float* xs = xh + (size_t)s * 384;
  float a0 = 0.0f, a1 = 0.0f;
#pragma unroll
  for (int h = 0; h < 6; ++h) {
    float w = alpha[(size_t)e * 6 + h] / zseg[t * 6 + h] * (1.0f / 6.0f);
    a0 += w * xs[h * 64 + lane];
    a1 += w * xs[h * 64 + lane + 32];
  }
  atomicAdd(&xout[(size_t)t * 64 + lane],      a0);
  atomicAdd(&xout[(size_t)t * 64 + lane + 32], a1);
}

// ---- fitness ----------------------------------------------------------------

__global__ void k_bvec(const float* __restrict__ xout, const float* __restrict__ lin2w,
                       const float* __restrict__ lin2b, float* bvec, float* fitsum, int N) {
  int i = blockIdx.x * blockDim.x + threadIdx.x;
  if (i >= N) return;
  float sacc = 0.0f;
  for (int c = 0; c < 64; ++c) sacc += xout[(size_t)i * 64 + c] * lin2w[c];
  bvec[i] = sacc + lin2b[0];
  fitsum[i] = 0.0f;
}
__global__ void k_rownorm(float* a, int N) {
  int i = blockIdx.x * blockDim.x + threadIdx.x;
  if (i >= N) return;
  float ss = 0.0f;
  for (int c = 0; c < 64; ++c) { float v = a[(size_t)i * 64 + c]; ss += v * v; }
  float inv = 1.0f / fmaxf(sqrtf(ss), 1e-8f);
  for (int c = 0; c < 64; ++c) a[(size_t)i * 64 + c] *= inv;
}
// wave per edge: fitsum[t] += b[t] * dot(a_n[t], a_n[s])
__global__ void k_fit(const int* __restrict__ src, const int* __restrict__ dst,
                      const float* __restrict__ an, const float* __restrict__ bvec,
                      float* fitsum, int E, int N) {
  int e = blockIdx.x * (blockDim.x >> 5) + (threadIdx.x >> 5);
  int lane = threadIdx.x & 31;
  if (e >= E + N) return;
  int t, s;
  if (e < E) { t = src[e]; s = dst[e]; } else { t = s = e - E; }
  float p = an[(size_t)t * 64 + lane]      * an[(size_t)s * 64 + lane]
          + an[(size_t)t * 64 + lane + 32] * an[(size_t)s * 64 + lane + 32];
#pragma unroll
  for (int o = 16; o > 0; o >>= 1) p += __shfl_xor(p, o, 32);
  if (lane == 0) atomicAdd(&fitsum[t], bvec[t] * p);
}
__global__ void k_sigmoid(const float* __restrict__ fitsum, float* fit, int N) {
  int i = blockIdx.x * blockDim.x + threadIdx.x;
  if (i < N) fit[i] = 1.0f / (1.0f + expf(-fitsum[i]));
}

// ---- launch -----------------------------------------------------------------

extern "C" void kernel_launch(void* const* d_in, const int* in_sizes, int n_in,
                              void* d_out, int out_size, void* d_ws, size_t ws_size,
                              hipStream_t stream) {
  const float* x     = (const float*)d_in[0];
  const int*   eidx  = (const int*)d_in[1];
  const float* Wgcn  = (const float*)d_in[2];
  const float* bgcn  = (const float*)d_in[3];
  const float* weight= (const float*)d_in[4];
  const float* att   = (const float*)d_in[5];
  const float* lin1  = (const float*)d_in[6];
  const float* lin2w = (const float*)d_in[7];
  const float* lin2b = (const float*)d_in[8];

  const int N = in_sizes[0] / 64;
  const int E = in_sizes[1] / 2;
  const int ET = E + N;
  const int* src = eidx;
  const int* dst = eidx + E;

  float* xout = (float*)d_out;                    // N*64 (accumulated here)
  float* fit  = xout + (size_t)N * 64;            // N

  // workspace layout (~193 MB for N=50k, E=400k); buffers reused once dead:
  float* ws = (float*)d_ws;
  size_t o = 0;
  float* dinv  = ws + o; o += (size_t)N;          // deg -> dinv; later bvec
  float* xw    = ws + o; o += (size_t)N * 64;     // xw; later 'a' matrix
  float* xp    = ws + o; o += (size_t)N * 64;     // x_pool; later fitness accum
  float* xh    = ws + o; o += (size_t)N * 384;
  float* xjm   = ws + o; o += (size_t)N * 384;
  float* mseg  = ws + o; o += (size_t)N * 6;
  float* zseg  = ws + o; o += (size_t)N * 6;
  float* alpha = ws + o; o += (size_t)ET * 6;
  float* bvec   = dinv;
  float* amat   = xw;
  float* fitsum = xp;

  const int T = 256;
  const int gN    = (N + T - 1) / T;
  const int gE    = (E + T - 1) / T;
  const int gNC   = (N * 64 + T - 1) / T;
  const int gNH   = (N * 384 + T - 1) / T;
  const int gN6   = (N * 6 + T - 1) / T;
  const int gET6  = (ET * 6 + T - 1) / T;
  const int wE    = (E + 7) / 8;                  // wave-per-edge, 8 waves/block
  const int wET   = (ET + 7) / 8;
  const int gTile = (N + 15) / 16;

  // 1) degree -> dinv
  k_init_deg   <<<gN, T, 0, stream>>>(dinv, N);
  k_deg_accum  <<<gE, T, 0, stream>>>(dst, dinv, E);
  k_deg_to_dinv<<<gN, T, 0, stream>>>(dinv, N);
  // 2) xw = x @ W_gcn + b   (WMMA, B via TDM->LDS)
  k_gemm64<<<gTile, 128, 0, stream>>>(x, Wgcn, bgcn, xw, N, 1);
  // 3) GCN aggregation
  k_init_xpool   <<<gNC, T, 0, stream>>>(dinv, xw, xp, xout, N);
  k_scatter_xpool<<<wE, T, 0, stream>>>(src, dst, dinv, xw, xp, E);
  // 4) xh = x_pool @ weight  (WMMA, K-slabs of B via TDM->LDS)
  k_gemm384<<<gTile, 256, 0, stream>>>(xp, weight, xh, N);
  // 5) attention
  k_init_xjmax<<<gNH, T, 0, stream>>>(xh, xjm, N);
  k_init_seg  <<<gN6, T, 0, stream>>>(mseg, zseg, N);
  k_segmax    <<<wE, T, 0, stream>>>(src, dst, xh, xjm, E);
  k_alpha     <<<wET, T, 0, stream>>>(src, dst, xh, xjm, att, alpha, mseg, E, N);
  k_exp       <<<gET6, T, 0, stream>>>(src, alpha, mseg, zseg, E, N);
  k_xout      <<<wET, T, 0, stream>>>(src, dst, xh, alpha, zseg, xout, E, N);
  // 6) fitness head
  k_gemm64 <<<gTile, 128, 0, stream>>>(xout, lin1, nullptr, amat, N, 0);
  k_bvec   <<<gN, T, 0, stream>>>(xout, lin2w, lin2b, bvec, fitsum, N);
  k_rownorm<<<gN, T, 0, stream>>>(amat, N);
  k_fit    <<<wET, T, 0, stream>>>(src, dst, amat, bvec, fitsum, E, N);
  k_sigmoid<<<gN, T, 0, stream>>>(fitsum, fit, N);
}
